// DCNv2PackFlowGuided_23021024706966
// MI455X (gfx1250) — matrix operand order repaired and checked by reading, not compiled, not run on hardware
//
#include <hip/hip_runtime.h>
#include <hip/hip_bf16.h>
#include <math.h>

// MI455X / gfx1250 flow-guided DCNv2.
// - All GEMM work (4 convs + deformable contraction) on v_wmma_f32_16x16x32_f16.
// - Conv activations staged into LDS via CDNA5 async-to-LDS copies
//   (global_load_async_to_lds_b128 + s_wait_asynccnt), halo zero-filled once,
//   so the WMMA inner loop is branch-free ds_load_b128 + wmma.
// Workspace requirement: ~96 MB (see offsets in kernel_launch).

typedef __attribute__((ext_vector_type(16))) _Float16 v16h;
typedef __attribute__((ext_vector_type(8)))  _Float16 v8h;
typedef __attribute__((ext_vector_type(8)))  float    v8f;

#define B_   4
#define H_   128
#define W_   128
#define DG_  16
#define K_   9
#define CP1  160   // conv1 input channels padded 130 -> 160 (5 chunks of 32)

#define USE_ASYNC_LDS 1   // flip to 0 if the async asm fails to assemble

static __device__ __forceinline__ void wait_async_lds() {
#if __has_builtin(__builtin_amdgcn_s_wait_asynccnt)
  __builtin_amdgcn_s_wait_asynccnt(0);
#else
  asm volatile("s_wait_asynccnt 0" ::: "memory");
#endif
}

// ---------------------------------------------------------------------------
// Pack [x_flow_warped(64) | x_current(64) | flow(2) | zeros(30)] into NHWC f16
// ---------------------------------------------------------------------------
__global__ __launch_bounds__(256) void pack_input_f16(
    const float* __restrict__ xfw, const float* __restrict__ xcur,
    const float* __restrict__ flow, _Float16* __restrict__ xcat) {
  int idx = blockIdx.x * 256 + threadIdx.x;
  const int total = B_ * H_ * W_ * CP1;
  if (idx >= total) return;
  int c = idx % CP1;
  int x = (idx / CP1) % W_;
  int y = (idx / (CP1 * W_)) % H_;
  int b =  idx / (CP1 * W_ * H_);
  int sp = y * W_ + x;
  float v = 0.f;
  if (c < 64)        v = xfw [(b * 64 + c) * H_ * W_ + sp];
  else if (c < 128)  v = xcur[(b * 64 + (c - 64)) * H_ * W_ + sp];
  else if (c == 128) v = flow[(b * 2 + 0) * H_ * W_ + sp];
  else if (c == 129) v = flow[(b * 2 + 1) * H_ * W_ + sp];
  xcat[idx] = (_Float16)v;
}

// ---------------------------------------------------------------------------
// Pre-pack OIHW f32 conv weights into WMMA A-fragment order (f16).
// A 16x32 order used here (contiguous k, same as B): lane L holds M = L%16; element e (0..15) holds
//   K = e + (L>=16 ? 16 : 0)
// Fragment index: (((mtile*9 + tap)*chunks + chunk)*32 + lane)*16 + e
// ---------------------------------------------------------------------------
__global__ __launch_bounds__(256) void pack_conv_wfrag(
    const float* __restrict__ w, _Float16* __restrict__ frag,
    int Mtiles, int cinReal, int chunks) {
  int idx = blockIdx.x * 256 + threadIdx.x;
  int total = Mtiles * 9 * chunks * 512;
  if (idx >= total) return;
  int e     =  idx & 15;
  int lane  = (idx >> 4) & 31;
  int chunk = (idx >> 9) % chunks;
  int tap   = (idx / (512 * chunks)) % 9;
  int mt    =  idx / (512 * chunks * 9);
  int kel   = e + ((lane >= 16) ? 16 : 0);
  int cin   = chunk * 32 + kel;
  int cout  = mt * 16 + (lane & 15);
  _Float16 v = (_Float16)0.f;
  if (cin < cinReal) v = (_Float16)w[(cout * cinReal + cin) * 9 + tap];
  frag[idx] = v;
}

// DCN weights: contraction index kk = cin*9 + k (matches einsum 'bckhw,ock').
__global__ __launch_bounds__(256) void pack_dcn_wfrag(
    const float* __restrict__ w, _Float16* __restrict__ frag) {
  int idx = blockIdx.x * 256 + threadIdx.x;
  const int total = 4 * 18 * 512;          // Mtiles=4, K=576 -> 18 chunks
  if (idx >= total) return;
  int e     =  idx & 15;
  int lane  = (idx >> 4) & 31;
  int chunk = (idx >> 9) % 18;
  int mt    =  idx / (512 * 18);
  int kel   = e + ((lane >= 16) ? 16 : 0);
  int kk    = chunk * 32 + kel;
  int cin   = kk / 9, kpos = kk % 9;
  int cout  = mt * 16 + (lane & 15);
  frag[idx] = (_Float16)w[(cout * 64 + cin) * 9 + kpos];
}

// ---------------------------------------------------------------------------
// Implicit-GEMM 3x3 SAME conv on WMMA with async-LDS activation staging.
// Block = 256 threads (8 waves) covers 32 output columns in one row:
//   wave w -> sub = w&1 (which 16-col half), mslice = w>>1 (cout 16-tile).
// Stage rows y-1..y+1, cols x0-1..x0+32 (34 px) x CP channels into LDS via
// global_load_async_to_lds_b128 (halo lanes ds_store zeros), then the inner
// loop is 9 taps x CHUNKS chunks of branch-free ds_load + wmma.
// ---------------------------------------------------------------------------
template<int CHUNKS>
__global__ __launch_bounds__(256) void conv3x3_wmma(
    const _Float16* __restrict__ in,
    const _Float16* __restrict__ wfrag,
    const float*    __restrict__ bias,
    _Float16* __restrict__ out, int cpOut,
    int Mtiles, int doLrelu) {
  constexpr int CP   = CHUNKS * 32;      // channels (padded)
  constexpr int SEGPP = CP / 8;          // 16-byte segments per pixel
  constexpr int NSEG  = 3 * 34 * SEGPP;  // total staging segments
  __shared__ __align__(32) _Float16 sIn[3 * 34 * CP];

  int tid  = threadIdx.x;
  int tile = blockIdx.x;                 // 0..2047 : 32-col tiles
  int b    = tile >> 9;                  // 512 tiles per batch image
  int rem  = tile & 511;
  int y    = rem >> 2;
  int x0   = (rem & 3) << 5;             // 32-col base

  // ---- stage halo tile into LDS (async) ----
  for (int s = tid; s < NSEG; s += 256) {
    int cseg = s % SEGPP;
    int px   = (s / SEGPP) % 34;         // 0..33  -> x = x0-1+px
    int row  = s / (SEGPP * 34);         // 0..2   -> y-1+row
    int xx = x0 - 1 + px;
    int yy = y - 1 + row;
    int hidx = (row * 34 + px) * CP + cseg * 8;   // half index in LDS
    bool valid = (yy >= 0) && (yy < H_) && (xx >= 0) && (xx < W_);
    if (valid) {
      const _Float16* gp = in + (size_t)((b * H_ + yy) * W_ + xx) * CP + cseg * 8;
#if USE_ASYNC_LDS
      uint32_t loff = (uint32_t)(uintptr_t)(&sIn[hidx]);
      uint64_t ga   = (uint64_t)(uintptr_t)gp;
      asm volatile("global_load_async_to_lds_b128 %0, %1, off"
                   :: "v"(loff), "v"(ga) : "memory");
#else
      *(v8h*)(&sIn[hidx]) = *(const v8h*)gp;
#endif
    } else {
      v8h z = {};
      *(v8h*)(&sIn[hidx]) = z;
    }
  }
#if USE_ASYNC_LDS
  wait_async_lds();
#endif
  __syncthreads();

  // ---- WMMA ----
  int wave = tid >> 5;
  int lane = tid & 31;
  int sub  = wave & 1;                   // which 16-col half
  int mt   = blockIdx.y * 4 + (wave >> 1);
  int n    = lane & 15;
  int hiL  = lane >> 4;

  if (mt < Mtiles) {                     // wave-uniform (after barrier)
    v8f acc = {};
    const _Float16* wbase = wfrag + (size_t)mt * 9 * CHUNKS * 512;
    #pragma unroll
    for (int tap = 0; tap < 9; ++tap) {
      int ky = tap / 3, kx = tap % 3;
      int pbase = (ky * 34 + sub * 16 + n + kx) * CP + hiL * 16;
      #pragma unroll
      for (int ch = 0; ch < CHUNKS; ++ch) {
        v16h a  = *(const v16h*)(wbase + ((size_t)(tap * CHUNKS + ch) * 32 + lane) * 16);
        v16h bf = *(const v16h*)(&sIn[pbase + ch * 32]);
        acc = __builtin_amdgcn_wmma_f32_16x16x32_f16(
            false, a, false, bf, (short)0, acc, false, false);
      }
    }
    // C/D layout: vgpr j -> cout = mt*16 + j + (lane>=16 ? 8 : 0); N = lane%16
    int coutBase = mt * 16 + hiL * 8;
    v8h o;
    #pragma unroll
    for (int j = 0; j < 8; ++j) {
      float v = acc[j] + bias[coutBase + j];
      if (doLrelu) v = (v >= 0.f) ? v : 0.1f * v;
      o[j] = (_Float16)v;
    }
    int x = x0 + sub * 16 + n;
    *(v8h*)(out + (size_t)((b * H_ + y) * W_ + x) * cpOut + coutBase) = o;
  }
}

// ---------------------------------------------------------------------------
// Deformable conv. Block = 128 threads (4 waves) handles 16 pixels.
// Stage 1: fuse tanh/sigmoid/flow-add, bilinear-sample x, mask-weight,
//          stage 16px x 576(K = cin*9+k) f16 tile in LDS.
// Stage 2: 4 waves (one cout 16-tile each) run 18 WMMA steps against
//          pre-packed DCN weight fragments; write NCHW f32 + bias.
// ---------------------------------------------------------------------------
__global__ __launch_bounds__(128) void dcn_wmma(
    const float* __restrict__ xin,      // (4,64,128,128) f32
    const float* __restrict__ flow,     // (4,2,128,128)  f32
    const _Float16* __restrict__ c4,    // NHWC (4,128,128,432) raw conv4 out
    const _Float16* __restrict__ wfragD,
    const float* __restrict__ dcnb,
    float* __restrict__ dout) {
  __shared__ __align__(32) _Float16 smp[16][576];   // 18 KB
  int tid = threadIdx.x;
  int ntile = blockIdx.x;               // 0..4095
  int b   = ntile >> 10;
  int rem = ntile & 1023;
  int y   = rem >> 3;
  int x0  = (rem & 7) << 4;

  for (int t = tid; t < 16 * DG_ * K_; t += 128) {   // 2304 / 128 = 18 exact
    int px = t / (DG_ * K_);
    int r2 = t % (DG_ * K_);
    int dg = r2 / K_;
    int k  = r2 % K_;
    int x  = x0 + px;
    int base = ((b * H_ + y) * W_ + x) * 432;
    // offset channel = dg*18 + k*2 + comp ; mask channel = 288 + dg*9 + k
    float oy = 10.f * tanhf((float)c4[base + dg * 18 + k * 2 + 0]);
    float ox = 10.f * tanhf((float)c4[base + dg * 18 + k * 2 + 1]);
    float mk = 1.f / (1.f + expf(-(float)c4[base + 288 + dg * 9 + k]));
    int sp = y * W_ + x;
    float fy = flow[(b * 2 + 1) * H_ * W_ + sp];   // flow_yx[0] = flow ch1
    float fx = flow[(b * 2 + 0) * H_ * W_ + sp];   // flow_yx[1] = flow ch0
    float ys = (float)y + (float)(k / 3 - 1) + oy + fy;
    float xs = (float)x + (float)(k % 3 - 1) + ox + fx;
    float yf = floorf(ys), xf = floorf(xs);
    float wy = ys - yf, wx = xs - xf;
    int yi = (int)yf, xi = (int)xf;
    float w00 = (1.f - wy) * (1.f - wx) * mk;
    float w01 = (1.f - wy) * wx * mk;
    float w10 = wy * (1.f - wx) * mk;
    float w11 = wy * wx * mk;
    #pragma unroll
    for (int cg = 0; cg < 4; ++cg) {
      int c = dg * 4 + cg;
      const float* img = xin + (size_t)(b * 64 + c) * H_ * W_;
      float v00 = 0.f, v01 = 0.f, v10 = 0.f, v11 = 0.f;
      if (yi >= 0 && yi < H_) {
        if (xi >= 0     && xi < W_)     v00 = img[yi * W_ + xi];
        if (xi + 1 >= 0 && xi + 1 < W_) v01 = img[yi * W_ + xi + 1];
      }
      if (yi + 1 >= 0 && yi + 1 < H_) {
        if (xi >= 0     && xi < W_)     v10 = img[(yi + 1) * W_ + xi];
        if (xi + 1 >= 0 && xi + 1 < W_) v11 = img[(yi + 1) * W_ + xi + 1];
      }
      smp[px][c * 9 + k] =
          (_Float16)(v00 * w00 + v01 * w01 + v10 * w10 + v11 * w11);
    }
  }
  __syncthreads();

  int wave = tid >> 5;                  // mtile 0..3
  int lane = tid & 31;
  int n = lane & 15, hiL = lane >> 4;
  v8f acc = {};
  const _Float16* wbase = wfragD + (size_t)wave * 18 * 512;
  #pragma unroll
  for (int ch = 0; ch < 18; ++ch) {
    v16h a  = *(const v16h*)(wbase + ((size_t)ch * 32 + lane) * 16);
    v16h bf = *(const v16h*)(&smp[n][ch * 32 + hiL * 16]);
    acc = __builtin_amdgcn_wmma_f32_16x16x32_f16(
        false, a, false, bf, (short)0, acc, false, false);
  }
  int coutBase = wave * 16 + hiL * 8;
  int x = x0 + n;
  #pragma unroll
  for (int j = 0; j < 8; ++j) {
    int co = coutBase + j;
    dout[((size_t)(b * 64 + co) * H_ + y) * W_ + x] = acc[j] + dcnb[co];
  }
}

// ---------------------------------------------------------------------------
extern "C" void kernel_launch(void* const* d_in, const int* in_sizes, int n_in,
                              void* d_out, int out_size, void* d_ws, size_t ws_size,
                              hipStream_t stream) {
  (void)in_sizes; (void)n_in; (void)out_size; (void)ws_size;
  const float* x    = (const float*)d_in[0];
  const float* xfw  = (const float*)d_in[1];
  const float* xcur = (const float*)d_in[2];
  const float* flow = (const float*)d_in[3];
  const float* w1   = (const float*)d_in[4];
  const float* b1   = (const float*)d_in[5];
  const float* w2   = (const float*)d_in[6];
  const float* b2   = (const float*)d_in[7];
  const float* w3   = (const float*)d_in[8];
  const float* b3   = (const float*)d_in[9];
  const float* w4   = (const float*)d_in[10];
  const float* b4   = (const float*)d_in[11];
  const float* dw   = (const float*)d_in[12];
  const float* db   = (const float*)d_in[13];
  float* out = (float*)d_out;

  // Workspace layout (bytes, 256-aligned); total ~95.3 MB
  char* ws = (char*)d_ws;
  _Float16* xcat = (_Float16*)(ws + 0);          // 20,971,520  (reused as h3)
  _Float16* h1   = (_Float16*)(ws + 20971520);   //  8,388,608
  _Float16* h2   = (_Float16*)(ws + 29360128);   //  8,388,608
  _Float16* c4o  = (_Float16*)(ws + 37748736);   // 56,623,104
  _Float16* wf1  = (_Float16*)(ws + 94371840);   //    184,320
  _Float16* wf2  = (_Float16*)(ws + 94556160);   //     73,728
  _Float16* wf3  = (_Float16*)(ws + 94629888);   //     73,728
  _Float16* wf4  = (_Float16*)(ws + 94703616);   //    497,664
  _Float16* wfd  = (_Float16*)(ws + 95201280);   //     73,728
  _Float16* h3   = xcat;                         // xcat dead after conv1

  // Packing
  hipLaunchKernelGGL(pack_input_f16, dim3((B_*H_*W_*CP1 + 255) / 256), dim3(256),
                     0, stream, xfw, xcur, flow, xcat);
  hipLaunchKernelGGL(pack_conv_wfrag, dim3((4*9*5*512 + 255) / 256), dim3(256),
                     0, stream, w1, wf1, 4, 130, 5);
  hipLaunchKernelGGL(pack_conv_wfrag, dim3((4*9*2*512 + 255) / 256), dim3(256),
                     0, stream, w2, wf2, 4, 64, 2);
  hipLaunchKernelGGL(pack_conv_wfrag, dim3((4*9*2*512 + 255) / 256), dim3(256),
                     0, stream, w3, wf3, 4, 64, 2);
  hipLaunchKernelGGL(pack_conv_wfrag, dim3((27*9*2*512 + 255) / 256), dim3(256),
                     0, stream, w4, wf4, 27, 64, 2);
  hipLaunchKernelGGL(pack_dcn_wfrag, dim3((4*18*512 + 255) / 256), dim3(256),
                     0, stream, dw, wfd);

  // Conv chain: grid.x = 2048 (32-col tiles), grid.y = cout-tile groups of 4
  hipLaunchKernelGGL((conv3x3_wmma<5>), dim3(2048, 1), dim3(256), 0, stream,
                     xcat, wf1, b1, h1, 64, 4, 1);
  hipLaunchKernelGGL((conv3x3_wmma<2>), dim3(2048, 1), dim3(256), 0, stream,
                     h1, wf2, b2, h2, 64, 4, 1);
  hipLaunchKernelGGL((conv3x3_wmma<2>), dim3(2048, 1), dim3(256), 0, stream,
                     h2, wf3, b3, h3, 64, 4, 1);
  hipLaunchKernelGGL((conv3x3_wmma<2>), dim3(2048, 7), dim3(256), 0, stream,
                     h3, wf4, b4, c4o, 432, 27, 0);

  // Deformable conv (tanh/sigmoid/flow fused into sampling stage)
  hipLaunchKernelGGL(dcn_wmma, dim3(4096), dim3(128), 0, stream,
                     x, flow, c4o, wfd, db, out);
}
